// GATv2WithGlobal_3143916060991
// MI455X (gfx1250) — matrix operand order repaired and checked by reading, not compiled, not run on hardware
//
#include <hip/hip_runtime.h>
#include <hip/hip_bf16.h>

// ---------------------------------------------------------------------------
// Types for CDNA5 WMMA (wave32)
// ---------------------------------------------------------------------------
typedef __attribute__((ext_vector_type(16))) __bf16 v16bf;
typedef __attribute__((ext_vector_type(8)))  __bf16 v8bf;
typedef __attribute__((ext_vector_type(8)))  float  v8f;

#define HC 128   // H*C
#define HEADS 4
#define CPH 32   // channels per head
#define MH 256
#define GF 32

// f32 -> bf16 (round to nearest even), returned as raw bits
__device__ __forceinline__ unsigned short f2bf(float f) {
    unsigned u = __float_as_uint(f);
    unsigned r = u + 0x7FFFu + ((u >> 16) & 1u);
    return (unsigned short)(r >> 16);
}

// monotone float <-> uint key for atomicMax-based segment max
__device__ __forceinline__ unsigned fkey(float f) {
    unsigned b = __float_as_uint(f);
    return (b & 0x80000000u) ? ~b : (b | 0x80000000u);
}
__device__ __forceinline__ float fdec(unsigned k) {
    unsigned b = (k & 0x80000000u) ? (k ^ 0x80000000u) : ~k;
    return __uint_as_float(b);
}

// ---------------------------------------------------------------------------
// 1. Pre-shuffle a 128x128 f32 weight into bf16 B-fragments matching the ISA
//    16-bit B 32x16 layout: lanes 0-15 hold K=0..7 then 16..23 (col = lane),
//    lanes 16-31 hold K=8..15 then 24..31 (col = lane-16), 2 bf16 per VGPR.
//    frag index = ((kt*8+nt)*512 + lane*16 + e)
// ---------------------------------------------------------------------------
__global__ __launch_bounds__(256) void build_wfrag(
    const float* __restrict__ W, unsigned short* __restrict__ frag) {
    int idx = blockIdx.x * blockDim.x + threadIdx.x;
    if (idx >= 128 * 128) return;
    int fragId = idx >> 9;          // kt*8 + nt
    int r      = idx & 511;
    int lane   = r >> 4;
    int e      = r & 15;
    int kt     = fragId >> 3;
    int nt     = fragId & 7;
    int col    = nt * 16 + (lane & 15);
    int k      = kt * 32 + ((lane >> 4) * 8) + (e < 8 ? e : 16 + (e - 8));
    frag[idx]  = f2bf(W[k * 128 + col]);
}

// ---------------------------------------------------------------------------
// 2. Node transform via WMMA bf16: X[N,128] -> xl = X@Wl+bl, xr = X@Wr+br
//    block = 256 thr (8 waves); block owns a 16-row tile; wave w owns col
//    tile nt=w; K=128 -> 4 wmma steps per accumulator.
// ---------------------------------------------------------------------------
__global__ __launch_bounds__(256) void node_transform(
    const float* __restrict__ X,
    const unsigned short* __restrict__ fragL,
    const unsigned short* __restrict__ fragR,
    const float* __restrict__ biasL, const float* __restrict__ biasR,
    float* __restrict__ xl, float* __restrict__ xr, int N) {
    __shared__ __align__(16) unsigned short tileA[16 * 128];  // 4 KB bf16

    int m0  = blockIdx.x * 16;
    int tid = threadIdx.x;

    // load 16x128 f32 tile, convert to bf16 in LDS (row-major)
#pragma unroll
    for (int j = 0; j < 8; ++j) {
        int idx = tid * 8 + j;            // 0..2047
        int row = idx >> 7, c = idx & 127;
        int gr  = m0 + row;
        float f = (gr < N) ? X[(size_t)gr * 128 + c] : 0.0f;
        tileA[idx] = f2bf(f);
    }
    __syncthreads();

    int lane = tid & 31;
    int nt   = tid >> 5;                  // column tile 0..7
    int row  = lane & 15;
    int half = lane >> 4;

    v8f accL = {};
    v8f accR = {};

#pragma unroll
    for (int kt = 0; kt < 4; ++kt) {
        // A fragment: 16-bit A 16x32 layout. half 0: K 0-7 then 16-23;
        // half 1: K 8-15 then 24-31.
        int aoff  = row * 128 + kt * 32 + half * 8;
        v8bf alo  = *(const v8bf*)&tileA[aoff];
        v8bf ahi  = *(const v8bf*)&tileA[aoff + 16];
        v16bf A;
#pragma unroll
        for (int i = 0; i < 8; ++i) { A[i] = alo[i]; A[i + 8] = ahi[i]; }

        // B fragments (pre-shuffled): 32 bytes per lane, contiguous
        size_t boff = ((size_t)(kt * 8 + nt)) * 512 + (size_t)lane * 16;
        v8bf bLlo = *(const v8bf*)&fragL[boff];
        v8bf bLhi = *(const v8bf*)&fragL[boff + 8];
        v8bf bRlo = *(const v8bf*)&fragR[boff];
        v8bf bRhi = *(const v8bf*)&fragR[boff + 8];
        v16bf BL, BR;
#pragma unroll
        for (int i = 0; i < 8; ++i) {
            BL[i] = bLlo[i]; BL[i + 8] = bLhi[i];
            BR[i] = bRlo[i]; BR[i + 8] = bRhi[i];
        }

        accL = __builtin_amdgcn_wmma_f32_16x16x32_bf16(
            false, A, false, BL, (short)0, accL, false, false);
        accR = __builtin_amdgcn_wmma_f32_16x16x32_bf16(
            false, A, false, BR, (short)0, accR, false, false);
    }

    // C/D layout: lane col = lane%16; VGPR i -> M = i + (lane>=16 ? 8 : 0)
    int col   = lane & 15;
    int rbase = half * 8;
    int cc    = nt * 16 + col;
    float bLv = biasL[cc];
    float bRv = biasR[cc];
#pragma unroll
    for (int i = 0; i < 8; ++i) {
        int r = m0 + rbase + i;
        if (r < N) {
            xl[(size_t)r * 128 + cc] = accL[i] + bLv;
            xr[(size_t)r * 128 + cc] = accR[i] + bRv;
        }
    }
}

// ---------------------------------------------------------------------------
// 3. Edge attention logits: one wave per edge. Lane l covers channel l of
//    every head. alpha_raw[e,h] = sum_c att[h,c]*lrelu(xl[src,h,c]+xr[dst,h,c])
//    Segment max via atomicMax on monotone uint keys.
// ---------------------------------------------------------------------------
__global__ __launch_bounds__(256) void edge_alpha(
    const int* __restrict__ ei, int E, int Etot,
    const float* __restrict__ xl, const float* __restrict__ xr,
    const float* __restrict__ att,
    float* __restrict__ araw, unsigned* __restrict__ amax) {
    int wid  = (int)(((size_t)blockIdx.x * blockDim.x + threadIdx.x) >> 5);
    int lane = threadIdx.x & 31;
    if (wid >= Etot) return;
    int src, dst;
    if (wid < E) { src = ei[wid]; dst = ei[E + wid]; }
    else         { src = wid - E; dst = src; }

    float v[HEADS];
#pragma unroll
    for (int h = 0; h < HEADS; ++h) {
        float s  = xl[(size_t)src * 128 + h * CPH + lane] +
                   xr[(size_t)dst * 128 + h * CPH + lane];
        float lr = (s > 0.0f) ? s : 0.2f * s;
        v[h] = att[h * CPH + lane] * lr;
    }
#pragma unroll
    for (int h = 0; h < HEADS; ++h)
#pragma unroll
        for (int off = 16; off > 0; off >>= 1)
            v[h] += __shfl_xor(v[h], off, 32);

    if (lane < HEADS) {
        araw[(size_t)wid * HEADS + lane] = v[lane];
        atomicMax(&amax[(size_t)dst * HEADS + lane], fkey(v[lane]));
    }
}

// ---------------------------------------------------------------------------
// 4. Edge scatter: a = exp(raw - max); denom[dst,h] += a;
//    out[dst,:] += a * xl[src,:]  (f32 atomics, L2-resident 51 MB target)
// ---------------------------------------------------------------------------
__global__ __launch_bounds__(256) void edge_scatter(
    const int* __restrict__ ei, int E, int Etot,
    const float* __restrict__ xl,
    const float* __restrict__ araw, const unsigned* __restrict__ amax,
    float* __restrict__ denom, float* __restrict__ out) {
    int wid  = (int)(((size_t)blockIdx.x * blockDim.x + threadIdx.x) >> 5);
    int lane = threadIdx.x & 31;
    if (wid >= Etot) return;
    int src, dst;
    if (wid < E) { src = ei[wid]; dst = ei[E + wid]; }
    else         { src = wid - E; dst = src; }

    float a[HEADS];
#pragma unroll
    for (int h = 0; h < HEADS; ++h) {
        float m = fdec(amax[(size_t)dst * HEADS + h]);
        a[h] = __expf(araw[(size_t)wid * HEADS + h] - m);
    }
    if (lane < HEADS) atomicAdd(&denom[(size_t)dst * HEADS + lane], a[lane]);
#pragma unroll
    for (int h = 0; h < HEADS; ++h)
        atomicAdd(&out[(size_t)dst * 128 + h * CPH + lane],
                  a[h] * xl[(size_t)src * 128 + h * CPH + lane]);
}

// ---------------------------------------------------------------------------
// 5. Normalize by denom, add bias, accumulate BN sums (per-channel).
//    block = 128 thr (one per channel), each block walks 128 nodes.
// ---------------------------------------------------------------------------
__global__ __launch_bounds__(128) void finalize_bn(
    float* __restrict__ hb, const float* __restrict__ denom,
    const float* __restrict__ bias,
    float* __restrict__ bnsum, float* __restrict__ bnsq, int N) {
    int c  = threadIdx.x;
    int n0 = blockIdx.x * 128;
    float bi = bias[c];
    float s = 0.0f, q = 0.0f;
    for (int i = 0; i < 128; ++i) {
        int n = n0 + i;
        if (n >= N) break;
        float d = denom[(size_t)n * HEADS + (c >> 5)] + 1e-16f;
        float v = hb[(size_t)n * 128 + c] / d + bi;
        hb[(size_t)n * 128 + c] = v;
        s += v;
        q += v * v;
    }
    atomicAdd(&bnsum[c], s);
    atomicAdd(&bnsq[c], q);
}

// ---------------------------------------------------------------------------
// 6. BatchNorm (biased var) + ReLU, in place.
// ---------------------------------------------------------------------------
__global__ __launch_bounds__(256) void bn_apply(
    float* __restrict__ hb, const float* __restrict__ bnsum,
    const float* __restrict__ bnsq,
    const float* __restrict__ gam, const float* __restrict__ bet,
    float invN, size_t total) {
    size_t idx = (size_t)blockIdx.x * blockDim.x + threadIdx.x;
    if (idx >= total) return;
    int c = (int)(idx & 127);
    float mu  = bnsum[c] * invN;
    float var = bnsq[c] * invN - mu * mu;
    float v = (hb[idx] - mu) * rsqrtf(var + 1e-5f) * gam[c] + bet[c];
    hb[idx] = v > 0.0f ? v : 0.0f;
}

// ---------------------------------------------------------------------------
// 7. Global mean pool (segment sums via atomics; batch ids are sorted).
// ---------------------------------------------------------------------------
__global__ __launch_bounds__(128) void pool_kernel(
    const float* __restrict__ hb, const int* __restrict__ batch,
    float* __restrict__ pooled, float* __restrict__ cnt, int N) {
    int c  = threadIdx.x;
    int n0 = blockIdx.x * 128;
    for (int i = 0; i < 128; ++i) {
        int n = n0 + i;
        if (n >= N) break;
        int b = batch[n];
        atomicAdd(&pooled[(size_t)b * 128 + c], hb[(size_t)n * 128 + c]);
        if (c == 0) atomicAdd(&cnt[b], 1.0f);
    }
}

// ---------------------------------------------------------------------------
// 8. MLP head: z=[pooled/cnt, gfeat] (160) -> relu(z@W1+b1) (256) -> @W2+b2
//    one block (256 thr) per graph.
// ---------------------------------------------------------------------------
__global__ __launch_bounds__(256) void mlp_kernel(
    const float* __restrict__ pooled, const float* __restrict__ cnt,
    const float* __restrict__ gfeat,
    const float* __restrict__ W1, const float* __restrict__ b1,
    const float* __restrict__ W2, const float* __restrict__ b2,
    float* __restrict__ out) {
    __shared__ float z[HC + GF];
    __shared__ float red[MH];
    int g = blockIdx.x;
    int t = threadIdx.x;
    if (t < HC)            z[t] = pooled[(size_t)g * HC + t] / fmaxf(cnt[g], 1.0f);
    else if (t < HC + GF)  z[t] = gfeat[(size_t)g * GF + (t - HC)];
    __syncthreads();

    float acc = b1[t];
    for (int k = 0; k < HC + GF; ++k) acc += z[k] * W1[k * MH + t];
    acc = acc > 0.0f ? acc : 0.0f;
    red[t] = acc * W2[t];
    __syncthreads();
    for (int s = MH / 2; s > 0; s >>= 1) {
        if (t < s) red[t] += red[t + s];
        __syncthreads();
    }
    if (t == 0) out[g] = red[0] + b2[0];
}

// ---------------------------------------------------------------------------
// Launch
// ---------------------------------------------------------------------------
extern "C" void kernel_launch(void* const* d_in, const int* in_sizes, int n_in,
                              void* d_out, int out_size, void* d_ws, size_t ws_size,
                              hipStream_t stream) {
    const float* x     = (const float*)d_in[0];
    const int*   ei    = (const int*)  d_in[1];
    const int*   batch = (const int*)  d_in[2];
    const float* gfeat = (const float*)d_in[3];
    const float* Wl1 = (const float*)d_in[4];  const float* bl1 = (const float*)d_in[5];
    const float* Wr1 = (const float*)d_in[6];  const float* br1 = (const float*)d_in[7];
    const float* att1= (const float*)d_in[8];  const float* bias1=(const float*)d_in[9];
    const float* g1  = (const float*)d_in[10]; const float* be1 = (const float*)d_in[11];
    const float* Wl2 = (const float*)d_in[12]; const float* bl2 = (const float*)d_in[13];
    const float* Wr2 = (const float*)d_in[14]; const float* br2 = (const float*)d_in[15];
    const float* att2= (const float*)d_in[16]; const float* bias2=(const float*)d_in[17];
    const float* g2  = (const float*)d_in[18]; const float* be2 = (const float*)d_in[19];
    const float* Wfc1= (const float*)d_in[20]; const float* bfc1= (const float*)d_in[21];
    const float* Wfc2= (const float*)d_in[22]; const float* bfc2= (const float*)d_in[23];
    float* out = (float*)d_out;

    const int N    = in_sizes[0] / 128;
    const int E    = in_sizes[1] / 2;
    const int G    = in_sizes[3] / GF;
    const int Etot = E + N;

    // workspace carve-out
    char*  ws  = (char*)d_ws;
    size_t off = 0;
    auto alloc = [&](size_t bytes) -> void* {
        void* p = ws + off;
        off = (off + bytes + 255) & ~(size_t)255;
        return p;
    };
    unsigned short* fL1 = (unsigned short*)alloc(16384 * sizeof(unsigned short));
    unsigned short* fR1 = (unsigned short*)alloc(16384 * sizeof(unsigned short));
    unsigned short* fL2 = (unsigned short*)alloc(16384 * sizeof(unsigned short));
    unsigned short* fR2 = (unsigned short*)alloc(16384 * sizeof(unsigned short));
    float*    xlb   = (float*)   alloc((size_t)N * 128 * sizeof(float));
    float*    xrb   = (float*)   alloc((size_t)N * 128 * sizeof(float));
    float*    hb    = (float*)   alloc((size_t)N * 128 * sizeof(float));
    float*    araw  = (float*)   alloc((size_t)Etot * HEADS * sizeof(float));
    unsigned* amax  = (unsigned*)alloc((size_t)N * HEADS * sizeof(unsigned));
    float*    denom = (float*)   alloc((size_t)N * HEADS * sizeof(float));
    float*    bnsum = (float*)   alloc(128 * sizeof(float));
    float*    bnsq  = (float*)   alloc(128 * sizeof(float));
    float*    pooled= (float*)   alloc((size_t)G * 128 * sizeof(float));
    float*    cnt   = (float*)   alloc((size_t)G * sizeof(float));

    // weight fragment builds (bf16, WMMA B-layout)
    const int wthreads = 128 * 128;
    build_wfrag<<<(wthreads + 255) / 256, 256, 0, stream>>>(Wl1, fL1);
    build_wfrag<<<(wthreads + 255) / 256, 256, 0, stream>>>(Wr1, fR1);
    build_wfrag<<<(wthreads + 255) / 256, 256, 0, stream>>>(Wl2, fL2);
    build_wfrag<<<(wthreads + 255) / 256, 256, 0, stream>>>(Wr2, fR2);

    const int    mblocks    = (N + 15) / 16;
    const int    eblocks    = (Etot + 7) / 8;     // 8 waves / block
    const int    nblocks128 = (N + 127) / 128;
    const size_t total      = (size_t)N * 128;
    const int    tblocks    = (int)((total + 255) / 256);
    const float  invN       = 1.0f / (float)N;

    // ---------------- Layer 1 ----------------
    node_transform<<<mblocks, 256, 0, stream>>>(x, fL1, fR1, bl1, br1, xlb, xrb, N);
    hipMemsetAsync(hb,    0, (size_t)N * 128 * sizeof(float), stream);
    hipMemsetAsync(denom, 0, (size_t)N * HEADS * sizeof(float), stream);
    hipMemsetAsync(amax,  0, (size_t)N * HEADS * sizeof(unsigned), stream);
    hipMemsetAsync(bnsum, 0, 128 * sizeof(float), stream);
    hipMemsetAsync(bnsq,  0, 128 * sizeof(float), stream);
    edge_alpha  <<<eblocks, 256, 0, stream>>>(ei, E, Etot, xlb, xrb, att1, araw, amax);
    edge_scatter<<<eblocks, 256, 0, stream>>>(ei, E, Etot, xlb, araw, amax, denom, hb);
    finalize_bn <<<nblocks128, 128, 0, stream>>>(hb, denom, bias1, bnsum, bnsq, N);
    bn_apply    <<<tblocks, 256, 0, stream>>>(hb, bnsum, bnsq, g1, be1, invN, total);

    // ---------------- Layer 2 ----------------
    node_transform<<<mblocks, 256, 0, stream>>>(hb, fL2, fR2, bl2, br2, xlb, xrb, N);
    hipMemsetAsync(hb,    0, (size_t)N * 128 * sizeof(float), stream);
    hipMemsetAsync(denom, 0, (size_t)N * HEADS * sizeof(float), stream);
    hipMemsetAsync(amax,  0, (size_t)N * HEADS * sizeof(unsigned), stream);
    hipMemsetAsync(bnsum, 0, 128 * sizeof(float), stream);
    hipMemsetAsync(bnsq,  0, 128 * sizeof(float), stream);
    edge_alpha  <<<eblocks, 256, 0, stream>>>(ei, E, Etot, xlb, xrb, att2, araw, amax);
    edge_scatter<<<eblocks, 256, 0, stream>>>(ei, E, Etot, xlb, araw, amax, denom, hb);
    finalize_bn <<<nblocks128, 128, 0, stream>>>(hb, denom, bias2, bnsum, bnsq, N);
    bn_apply    <<<tblocks, 256, 0, stream>>>(hb, bnsum, bnsq, g2, be2, invN, total);

    // ---------------- Pool + MLP ----------------
    hipMemsetAsync(pooled, 0, (size_t)G * 128 * sizeof(float), stream);
    hipMemsetAsync(cnt,    0, (size_t)G * sizeof(float), stream);
    pool_kernel<<<nblocks128, 128, 0, stream>>>(hb, batch, pooled, cnt, N);
    mlp_kernel <<<G, 256, 0, stream>>>(pooled, cnt, gfeat, Wfc1, bfc1, Wfc2, bfc2, out);
}
